// GraphPhaseKuramoto_8117488189519
// MI455X (gfx1250) — matrix operand order repaired and checked by reading, not compiled, not run on hardware
//
#include <hip/hip_runtime.h>
#include <hip/hip_bf16.h>
#include <math.h>

// ---------------------------------------------------------------------------
// GraphPhaseKuramoto for MI455X (gfx1250, wave32, WMMA).
//
// Per step:  sum_j adj_ij sin(th_j - th_i - a_ij)
//              = cos(th_i)*S1_i - sin(th_i)*S2_i
//            S1 = P sin(th) - Q cos(th),  S2 = P cos(th) + Q sin(th)
//            P = adj .* cos(alpha), Q = adj .* sin(alpha)   (time invariant)
// => 60 steps of a [784 x 1568] x [1568 x 2] f16 GEMM per batch with
//    v_wmma_f32_16x16x32_f16.  P,Q in f16 = 157 MB -> resident in 192MB L2.
//
// Workspace layout (bytes, 256-aligned), total ~165 MB:
//   y1     : 64*16*784*4         = 3,211,264
//   stats  : 16*2*4 (pad 256)    @ 3,211,264
//   featn  : 64*784*8*4          @ 3,211,520
//   gamma  : 64*784*4            @ 4,817,152
//   skew   : 784*784*4           @ 5,017,856
//   amm    : 64*2*4 (pad 512)    @ 7,476,480
//   W(f16) : 64*49*49*512 halves @ 7,476,992   (157,351,936 B)
// ---------------------------------------------------------------------------

typedef __attribute__((ext_vector_type(16))) _Float16 v16h;
typedef __attribute__((ext_vector_type(8)))  float    v8f;

#define NNODES 784
#define NB 64
#define NMT 49                 // 784/16 M tiles
#define NKT 49                 // 1568/32 K tiles
#define WMT_HALVES 25088       // halves per (batch, mtile) strip = 49*512

// ------------------------------ K1: conv1 ----------------------------------
__global__ __launch_bounds__(256)
void k1_conv1(const float* __restrict__ x, const float* __restrict__ w1,
              const float* __restrict__ b1, float* __restrict__ y1) {
  int idx = blockIdx.x * 256 + threadIdx.x;
  if (idx >= NB * 16 * NNODES) return;
  int p = idx % NNODES;
  int c = (idx / NNODES) % 16;
  int b = idx / (NNODES * 16);
  int ph = p / 28, pw = p % 28;
  float acc = b1[c];
  for (int ci = 0; ci < 3; ++ci) {
    for (int kh = 0; kh < 3; ++kh) {
      int ih = ph + kh - 1;
      if (ih < 0 || ih >= 28) continue;
      for (int kw = 0; kw < 3; ++kw) {
        int iw = pw + kw - 1;
        if (iw < 0 || iw >= 28) continue;
        acc += x[(b * 3 + ci) * NNODES + ih * 28 + iw] *
               w1[((c * 3 + ci) * 3 + kh) * 3 + kw];
      }
    }
  }
  y1[(b * 16 + c) * NNODES + p] = acc;
}

// --------------------------- K2: BN batch stats ----------------------------
__global__ __launch_bounds__(256)
void k2_stats(const float* __restrict__ y1, float* __restrict__ stats) {
  int c = blockIdx.x, tid = threadIdx.x;
  __shared__ float ssum[256], ssq[256];
  float s = 0.f, q = 0.f;
  for (int i = tid; i < NB * NNODES; i += 256) {
    int b = i / NNODES, p = i - b * NNODES;
    float v = y1[(b * 16 + c) * NNODES + p];
    s += v; q += v * v;
  }
  ssum[tid] = s; ssq[tid] = q;
  __syncthreads();
  for (int st = 128; st > 0; st >>= 1) {
    if (tid < st) { ssum[tid] += ssum[tid + st]; ssq[tid] += ssq[tid + st]; }
    __syncthreads();
  }
  if (tid == 0) {
    float mean = ssum[0] / (float)(NB * NNODES);
    float var  = ssq[0] / (float)(NB * NNODES) - mean * mean;
    stats[2 * c]     = mean;
    stats[2 * c + 1] = rsqrtf(var + 1e-5f);
  }
}

// ---------- K3: BN+tanh + conv2+tanh + 1x1 heads (+feat normalize) ---------
__global__ __launch_bounds__(256)
void k3_backbone(const float* __restrict__ y1, const float* __restrict__ stats,
                 const float* __restrict__ bng, const float* __restrict__ bnb,
                 const float* __restrict__ w2, const float* __restrict__ b2,
                 const float* __restrict__ fw, const float* __restrict__ fb,
                 const float* __restrict__ dw, const float* __restrict__ db,
                 float* __restrict__ featn, float* __restrict__ gamma) {
  int b = blockIdx.x, tid = threadIdx.x;
  __shared__ float h1[16 * NNODES];
  __shared__ float sw2[2304], sfw[128], sdw[16], sb2[16], sfb[8], sdb[1];
  for (int i = tid; i < 2304; i += 256) sw2[i] = w2[i];
  if (tid < 128) sfw[tid] = fw[tid];
  if (tid < 16) { sdw[tid] = dw[tid]; sb2[tid] = b2[tid]; }
  if (tid < 8)  sfb[tid] = fb[tid];
  if (tid == 0) sdb[0] = db[0];
  for (int i = tid; i < 16 * NNODES; i += 256) {
    int c = i / NNODES, p = i - c * NNODES;
    float v = y1[(b * 16 + c) * NNODES + p];
    h1[i] = tanhf((v - stats[2 * c]) * stats[2 * c + 1] * bng[c] + bnb[c]);
  }
  __syncthreads();
  for (int p = tid; p < NNODES; p += 256) {
    int ph = p / 28, pw = p % 28;
    float fe[8];
    #pragma unroll
    for (int f = 0; f < 8; ++f) fe[f] = sfb[f];
    float g = sdb[0];
    for (int oc = 0; oc < 16; ++oc) {
      float acc = sb2[oc];
      for (int ic = 0; ic < 16; ++ic) {
        const float* wp = &sw2[(oc * 16 + ic) * 9];
        const float* hp = &h1[ic * NNODES];
        #pragma unroll
        for (int kh = 0; kh < 3; ++kh) {
          int ih = ph + kh - 1;
          if (ih < 0 || ih >= 28) continue;
          #pragma unroll
          for (int kw = 0; kw < 3; ++kw) {
            int iw = pw + kw - 1;
            if (iw < 0 || iw >= 28) continue;
            acc += wp[kh * 3 + kw] * hp[ih * 28 + iw];
          }
        }
      }
      float h2v = tanhf(acc);
      #pragma unroll
      for (int f = 0; f < 8; ++f) fe[f] += sfw[f * 16 + oc] * h2v;
      g += sdw[oc] * h2v;
    }
    float nr = 0.f;
    #pragma unroll
    for (int f = 0; f < 8; ++f) nr += fe[f] * fe[f];
    float inv = 1.0f / fmaxf(sqrtf(nr), 1e-12f);
    #pragma unroll
    for (int f = 0; f < 8; ++f) featn[((size_t)b * NNODES + p) * 8 + f] = fe[f] * inv;
    gamma[b * NNODES + p] = g;
  }
}

// ------------------------- K4: dir skew = tanh(D-D^T) ----------------------
__global__ __launch_bounds__(256)
void k4_skew(const float* __restrict__ dir, float* __restrict__ skew) {
  int idx = blockIdx.x * 256 + threadIdx.x;
  if (idx >= NNODES * NNODES) return;
  int i = idx / NNODES, j = idx - i * NNODES;
  skew[idx] = tanhf(dir[idx] - dir[j * NNODES + i]);
}

__device__ __forceinline__ float dot8(const float* a, const float* b) {
  float d = 0.f;
  #pragma unroll
  for (int q = 0; q < 8; ++q) d += a[q] * b[q];
  return d;
}

// --------------- K5: per-batch min/max of relu(adj) (incl diag) ------------
__global__ __launch_bounds__(256)
void k5_minmax(const float* __restrict__ featn, float* __restrict__ amm) {
  int b = blockIdx.x, tid = threadIdx.x;
  __shared__ float fn[NNODES * 8];
  __shared__ float rmn[256], rmx[256];
  for (int i = tid; i < NNODES * 8; i += 256) fn[i] = featn[(size_t)b * NNODES * 8 + i];
  __syncthreads();
  float lmin = 3.4e38f, lmax = 0.0f;
  for (int k = tid; k < NNODES * NNODES; k += 256) {
    int i = k / NNODES, j = k - i * NNODES;
    float adj = (i == j) ? 0.0f : dot8(&fn[i * 8], &fn[j * 8]);
    float rl = fmaxf(adj, 0.0f);
    lmin = fminf(lmin, rl);
    lmax = fmaxf(lmax, rl);
  }
  rmn[tid] = lmin; rmx[tid] = lmax;
  __syncthreads();
  for (int st = 128; st > 0; st >>= 1) {
    if (tid < st) {
      rmn[tid] = fminf(rmn[tid], rmn[tid + st]);
      rmx[tid] = fmaxf(rmx[tid], rmx[tid + st]);
    }
    __syncthreads();
  }
  if (tid == 0) { amm[2 * b] = rmn[0]; amm[2 * b + 1] = rmx[0]; }
}

// ------ K6: build W = interleaved (P,Q) f16 in WMMA A-fragment layout ------
// Layout: [b][mtile][ktile][lane][16 halves].  For lane L (L%16 = row r within
// the 16-row tile; L/16 = kgroup), half h maps to in-tile K offset
//   ko = h + 16*(L>>4)
// (contiguous K order, matching the B fragment in K7).  Global K index
// kk = ktile*32+ko; kk even -> P[i, kk/2], kk odd -> Q[i, kk/2].
__global__ __launch_bounds__(256)
void k6_buildW(const float* __restrict__ featn, const float* __restrict__ skew,
               const float* __restrict__ amm, _Float16* __restrict__ W) {
  int blk = blockIdx.x;              // b*49 + mtile
  int b = blk / NMT, mt = blk - b * NMT;
  int tid = threadIdx.x;
  __shared__ float fn[NNODES * 8];
  __shared__ float s_cmin, s_invd;
  for (int i = tid; i < NNODES * 8; i += 256) fn[i] = featn[(size_t)b * NNODES * 8 + i];
  if (tid == 0) {
    float amin = amm[2 * b], amax = amm[2 * b + 1];
    float c_min = 1.0f / (amax + 1e-6f);
    float c_max = 1.0f / (amin + 1e-6f);
    s_cmin = c_min;
    s_invd = 1.0f / (c_max - c_min + 1e-6f);
  }
  __syncthreads();
  float cmin = s_cmin, invd = s_invd;
  _Float16* Wmt = W + (size_t)blk * WMT_HALVES;
  for (int p = tid; p < WMT_HALVES / 2; p += 256) {
    int f = p << 1;                  // even half index: (P,Q) pair
    int kt   = f >> 9;
    int rem  = f & 511;
    int lane = rem >> 4;
    int h    = rem & 15;
    int r = lane & 15, g = lane >> 4;
    int ko = h + 16 * g;
    int kk = kt * 32 + ko;
    int t = kk >> 1;
    int i = mt * 16 + r;
    float adj = (i == t) ? 0.0f : dot8(&fn[i * 8], &fn[t * 8]);
    float rl = fmaxf(adj, 0.0f);
    float cost = 1.0f / (rl + 1e-6f);
    float nc = (cost - cmin) * invd;
    float al = 1.5f * skew[i * NNODES + t] * nc;
    float sa, ca;
    sincosf(al, &sa, &ca);
    union { _Float16 hh[2]; unsigned u; } pk;
    pk.hh[0] = (_Float16)(adj * ca);   // P
    pk.hh[1] = (_Float16)(adj * sa);   // Q
    *reinterpret_cast<unsigned*>(Wmt + f) = pk.u;
  }
}

// ------------------- K7: persistent Kuramoto stepping (WMMA) ---------------
__global__ __launch_bounds__(512)
void k7_kuramoto(const float* __restrict__ omega, const float* __restrict__ kappa,
                 const float* __restrict__ gamma, const float* __restrict__ theta0,
                 const _Float16* __restrict__ W, float* __restrict__ out) {
  const int b = blockIdx.x;
  const int tid = threadIdx.x;
  const int lane = tid & 31;
  const int wave = tid >> 5;         // 16 waves
  __shared__ float th[NNODES], sT[NNODES], cT[NNODES], S1[NNODES], S2[NNODES];
  __shared__ unsigned sc[NNODES];    // packed (f16 sin | f16 cos) per node

  const int i1 = tid + 512;
  float om0 = 0.f, om1 = 0.f, kp0 = 0.f, kp1 = 0.f, gm0 = 0.f, gm1 = 0.f;
  if (tid < NNODES) {
    om0 = omega[tid]; kp0 = kappa[tid]; gm0 = gamma[b * NNODES + tid];
    th[tid] = theta0[b * NNODES + tid];
  }
  if (i1 < NNODES) {
    om1 = omega[i1]; kp1 = kappa[i1]; gm1 = gamma[b * NNODES + i1];
    th[i1] = theta0[b * NNODES + i1];
  }

  const _Float16* Wb = W + (size_t)b * NMT * WMT_HALVES;
  const int col = lane & 15;
  const int kg  = lane >> 4;
  const int mt0 = wave, mt1 = wave + 16, mt2 = wave + 32, mt3 = 48;
  const _Float16* A0 = Wb + (size_t)mt0 * WMT_HALVES + lane * 16;
  const _Float16* A1 = Wb + (size_t)mt1 * WMT_HALVES + lane * 16;
  const _Float16* A2 = Wb + (size_t)mt2 * WMT_HALVES + lane * 16;
  const _Float16* A3 = Wb + (size_t)mt3 * WMT_HALVES + lane * 16;
  // wave index is uniform per wave: make the 4th-tile branch a scalar branch
  const bool has4 = (__builtin_amdgcn_readfirstlane(wave) == 0);

  for (int step = 0; step < 60; ++step) {
    __syncthreads();
    // ---- phase A: trig + packed (sin,cos) vector ----
    if (tid < NNODES) {
      float s, c; sincosf(th[tid], &s, &c);
      sT[tid] = s; cT[tid] = c;
      union { _Float16 hh[2]; unsigned u; } pk;
      pk.hh[0] = (_Float16)s; pk.hh[1] = (_Float16)c;
      sc[tid] = pk.u;
    }
    if (i1 < NNODES) {
      float s, c; sincosf(th[i1], &s, &c);
      sT[i1] = s; cT[i1] = c;
      union { _Float16 hh[2]; unsigned u; } pk;
      pk.hh[0] = (_Float16)s; pk.hh[1] = (_Float16)c;
      sc[i1] = pk.u;
    }
    __syncthreads();

    // ---- phase B: S = W_b * Bvec via WMMA, per-wave M tiles ----
    v8f a0 = {}, a1 = {}, a2 = {}, a3 = {};
    for (int kt = 0; kt < NKT; ++kt) {
      // Build B fragment.  Lane covers K = kt*32 + kg*16 + e (e=0..15),
      // i.e. nodes t0..t0+7 with t0 = kt*16 + kg*8, two K per node:
      //   col 0: (sin t, -cos t)   col 1: (cos t, sin t)   col>=2: 0
      union { unsigned u[8]; v16h v; } B;
      if (col < 2) {
        int t0 = kt * 16 + kg * 8;
        const uint4* sp = (const uint4*)&sc[t0];   // 16B-aligned (t0 % 8 == 0)
        uint4 lo = sp[0], hi = sp[1];
        if (col == 0) {            // (s, -c): flip sign of high half
          B.u[0] = lo.x ^ 0x80000000u; B.u[1] = lo.y ^ 0x80000000u;
          B.u[2] = lo.z ^ 0x80000000u; B.u[3] = lo.w ^ 0x80000000u;
          B.u[4] = hi.x ^ 0x80000000u; B.u[5] = hi.y ^ 0x80000000u;
          B.u[6] = hi.z ^ 0x80000000u; B.u[7] = hi.w ^ 0x80000000u;
        } else {                   // (c, s): rotate halves
          B.u[0] = (lo.x << 16) | (lo.x >> 16); B.u[1] = (lo.y << 16) | (lo.y >> 16);
          B.u[2] = (lo.z << 16) | (lo.z >> 16); B.u[3] = (lo.w << 16) | (lo.w >> 16);
          B.u[4] = (hi.x << 16) | (hi.x >> 16); B.u[5] = (hi.y << 16) | (hi.y >> 16);
          B.u[6] = (hi.z << 16) | (hi.z >> 16); B.u[7] = (hi.w << 16) | (hi.w >> 16);
        }
      } else {
        #pragma unroll
        for (int e = 0; e < 8; ++e) B.u[e] = 0u;
      }
      v16h bf = B.v;

      v16h fa0 = *(const v16h*)(A0 + kt * 512);
      a0 = __builtin_amdgcn_wmma_f32_16x16x32_f16(false, fa0, false, bf,
                                                  (short)0, a0, false, false);
      v16h fa1 = *(const v16h*)(A1 + kt * 512);
      a1 = __builtin_amdgcn_wmma_f32_16x16x32_f16(false, fa1, false, bf,
                                                  (short)0, a1, false, false);
      v16h fa2 = *(const v16h*)(A2 + kt * 512);
      a2 = __builtin_amdgcn_wmma_f32_16x16x32_f16(false, fa2, false, bf,
                                                  (short)0, a2, false, false);
      if (has4) {
        v16h fa3 = *(const v16h*)(A3 + kt * 512);
        a3 = __builtin_amdgcn_wmma_f32_16x16x32_f16(false, fa3, false, bf,
                                                    (short)0, a3, false, false);
      }
    }
    // D layout: VGPR r -> lanes 0-15: (M=r, N=lane); lanes 16-31: (M=r+8, N=lane-16)
    {
      int rb0 = mt0 * 16 + (kg ? 8 : 0);
      int rb1 = mt1 * 16 + (kg ? 8 : 0);
      int rb2 = mt2 * 16 + (kg ? 8 : 0);
      int rb3 = mt3 * 16 + (kg ? 8 : 0);
      if (col == 0) {
        #pragma unroll
        for (int r = 0; r < 8; ++r) {
          S1[rb0 + r] = a0[r]; S1[rb1 + r] = a1[r]; S1[rb2 + r] = a2[r];
        }
        if (has4) {
          #pragma unroll
          for (int r = 0; r < 8; ++r) S1[rb3 + r] = a3[r];
        }
      } else if (col == 1) {
        #pragma unroll
        for (int r = 0; r < 8; ++r) {
          S2[rb0 + r] = a0[r]; S2[rb1 + r] = a1[r]; S2[rb2 + r] = a2[r];
        }
        if (has4) {
          #pragma unroll
          for (int r = 0; r < 8; ++r) S2[rb3 + r] = a3[r];
        }
      }
    }
    __syncthreads();

    // ---- phase C: theta update ----
    if (tid < NNODES) {
      float t0 = th[tid];
      float inter = cT[tid] * S1[tid] - sT[tid] * S2[tid];
      float td = om0 + (18.0f / 784.0f) * inter + kp0 * sinf(gm0 - t0);
      th[tid] = t0 + 0.1f * td;
    }
    if (i1 < NNODES) {
      float t0 = th[i1];
      float inter = cT[i1] * S1[i1] - sT[i1] * S2[i1];
      float td = om1 + (18.0f / 784.0f) * inter + kp1 * sinf(gm1 - t0);
      th[i1] = t0 + 0.1f * td;
    }
  }
  __syncthreads();
  if (tid < NNODES) out[b * NNODES + tid] = th[tid];
  if (i1 < NNODES)  out[b * NNODES + i1]  = th[i1];
}

// ---------------------------------------------------------------------------
extern "C" void kernel_launch(void* const* d_in, const int* in_sizes, int n_in,
                              void* d_out, int out_size, void* d_ws, size_t ws_size,
                              hipStream_t stream) {
  const float* x       = (const float*)d_in[0];
  const float* conv1_w = (const float*)d_in[1];
  const float* conv1_b = (const float*)d_in[2];
  const float* bn_g    = (const float*)d_in[3];
  const float* bn_b    = (const float*)d_in[4];
  const float* conv2_w = (const float*)d_in[5];
  const float* conv2_b = (const float*)d_in[6];
  const float* feat_w  = (const float*)d_in[7];
  const float* feat_b  = (const float*)d_in[8];
  const float* drive_w = (const float*)d_in[9];
  const float* drive_b = (const float*)d_in[10];
  const float* omega   = (const float*)d_in[11];
  const float* kappa   = (const float*)d_in[12];
  const float* dirm    = (const float*)d_in[13];
  const float* theta0  = (const float*)d_in[14];

  char* ws = (char*)d_ws;
  float*    y1    = (float*)(ws + 0);
  float*    stats = (float*)(ws + 3211264);
  float*    featn = (float*)(ws + 3211520);
  float*    gamma = (float*)(ws + 4817152);
  float*    skew  = (float*)(ws + 5017856);
  float*    amm   = (float*)(ws + 7476480);
  _Float16* W     = (_Float16*)(ws + 7476992);   // 157,351,936 bytes

  k1_conv1<<<(NB * 16 * NNODES + 255) / 256, 256, 0, stream>>>(x, conv1_w, conv1_b, y1);
  k2_stats<<<16, 256, 0, stream>>>(y1, stats);
  k3_backbone<<<NB, 256, 0, stream>>>(y1, stats, bn_g, bn_b, conv2_w, conv2_b,
                                      feat_w, feat_b, drive_w, drive_b, featn, gamma);
  k4_skew<<<(NNODES * NNODES + 255) / 256, 256, 0, stream>>>(dirm, skew);
  k5_minmax<<<NB, 256, 0, stream>>>(featn, amm);
  k6_buildW<<<NB * NMT, 256, 0, stream>>>(featn, skew, amm, W);
  k7_kuramoto<<<NB, 512, 0, stream>>>(omega, kappa, gamma, theta0, W, (float*)d_out);
}
